// SpikeConv_69277822484883
// MI455X (gfx1250) — matrix operand order repaired and checked by reading, not compiled, not run on hardware
//
#include <hip/hip_runtime.h>

typedef __attribute__((ext_vector_type(16))) _Float16 v16h;
typedef __attribute__((ext_vector_type(8)))  float    v8f;
typedef __attribute__((ext_vector_type(4)))  int      v4i;

#define TSTRIDE    4194304   // B*C*H*W = 16*256*1024
#define IMG_STRIDE 262144    // C*H*W   = 256*1024
#define HW         1024
// Padded spike layout: SP2[n(64)][cs(8)][row(34)][col(40)][ci(32)] f16, halo = 0
#define CSP        43520     // halves per (n,cs) plane = 34*40*32
#define SPROWH     1280      // halves per padded row   = 40*32
#define SP_BYTES   (64ull*8ull*CSP*2ull)       // 44,564,480 B
#define WT_OFF     SP_BYTES                     // f16 [9][8][256][32] = 1,179,648 B
#define STAT_OFF   (WT_OFF + 1179648ull)        // mean[256], invstd[256]

// ---------------------------------------------------------------------------
// Async global->LDS 16B copy (signature confirmed by compiler diagnostics).
// ---------------------------------------------------------------------------
__device__ __forceinline__ void async_copy16(const _Float16* g, _Float16* l) {
#if __has_builtin(__builtin_amdgcn_global_load_async_to_lds_b128)
    __builtin_amdgcn_global_load_async_to_lds_b128(
        (__attribute__((address_space(1))) v4i*)const_cast<_Float16*>(g),
        (__attribute__((address_space(3))) v4i*)l, 0, 0);
#else
    *(uint4*)l = *(const uint4*)g;
#endif
}

__device__ __forceinline__ void async_join() {
#if __has_builtin(__builtin_amdgcn_s_wait_asynccnt)
    __builtin_amdgcn_s_wait_asynccnt(0);
#elif __has_builtin(__builtin_amdgcn_global_load_async_to_lds_b128)
    asm volatile("s_wait_asynccnt 0" ::: "memory");
#endif
}

// ---------------------------------------------------------------------------
// Kernel 0: zero the padded spike buffer (halo must be 0).
// ---------------------------------------------------------------------------
__global__ void zero_kernel(uint4* __restrict__ p) {
    uint4 z; z.x = 0; z.y = 0; z.z = 0; z.w = 0;
    p[(size_t)blockIdx.x * 256 + threadIdx.x] = z;
}

// ---------------------------------------------------------------------------
// Kernel 1: LIF scan over T=4 with LDS transpose into ci-innermost layout.
// Block = (b, cs, 128-pixel slab): 32 ci x 128 pix; 16 membrane states/thread.
// grid = 16 * 8 * 8 = 1024 blocks.
// ---------------------------------------------------------------------------
__global__ void lif_kernel(const float* __restrict__ x, _Float16* __restrict__ sp) {
    __shared__ alignas(16) _Float16 tile[128 * 32];   // [pix][ci], 8KB
    const int tid = threadIdx.x;
    const int pb  = blockIdx.x & 7;          // 4-row pixel slab
    const int cs  = (blockIdx.x >> 3) & 7;   // channel chunk
    const int b   = blockIdx.x >> 6;         // batch
    const int h0  = pb * 4;
    const int ci  = tid >> 3;                // 0..31
    const int pq  = tid & 7;                 // 16-pixel group

    const int c = cs * 32 + ci;
    const size_t xbase = ((size_t)b * 256 + c) * HW + h0 * 32 + pq * 16;

    float v[16];
#pragma unroll
    for (int j = 0; j < 16; ++j) v[j] = 0.0f;

#pragma unroll 1
    for (int t = 0; t < 4; ++t) {
        const float* xp = x + (size_t)t * TSTRIDE + xbase;
#pragma unroll
        for (int j = 0; j < 16; ++j) {
            float vt = 0.5f * (v[j] + xp[j]);
            float s  = (vt >= 1.0f) ? 1.0f : 0.0f;
            v[j] = vt * (1.0f - s);
            tile[(pq * 16 + j) * 32 + ci] = (_Float16)s;
        }
        __syncthreads();
        // coalesced copy-out: 512 x 16B segments, 2 per thread
        const int n = t * 16 + b;
        _Float16* op = sp + ((size_t)n * 8 + cs) * CSP;
#pragma unroll
        for (int q = 0; q < 2; ++q) {
            const int s      = tid + q * 256;
            const int row    = s >> 7;       // 0..3
            const int within = s & 127;
            const int col    = within >> 2;  // 0..31
            const int sg     = within & 3;
            *(uint4*)(op + (size_t)(h0 + row + 1) * SPROWH + (col + 1) * 32 + sg * 8)
                = *(const uint4*)&tile[(row * 32 + col) * 32 + sg * 8];
        }
        __syncthreads();
    }
}

// ---------------------------------------------------------------------------
// Kernel 2: weights f32 [co][ci][r] -> f16 Wt2[r][cs][co][ci32]
// so each (r, cs, co_base) B-tile (64 co x 32 ci) is a contiguous 4KB run.
// ---------------------------------------------------------------------------
__global__ void wt_kernel(const float* __restrict__ w, _Float16* __restrict__ wt) {
    const int o   = blockIdx.x * 256 + threadIdx.x;  // 0 .. 589823
    const int c32 = o & 31;
    const int co  = (o >> 5) & 255;
    const int cs  = (o >> 13) & 7;
    const int r   = o >> 16;
    const int ci  = cs * 32 + c32;
    wt[o] = (_Float16)w[co * 2304 + ci * 9 + r];
}

// ---------------------------------------------------------------------------
// Kernel 3: implicit-GEMM conv, v_wmma_f32_16x16x32_f16.
// Block = 256 threads (8 waves): 128-pixel x 64-co tile.
// Spike tile (ci-innermost) staged once per chunk; A frag = 2 ds_load_b128.
// ---------------------------------------------------------------------------
union Frag { uint4 u[2]; v16h v; };

__launch_bounds__(256)
__global__ void conv_wmma_kernel(const _Float16* __restrict__ sp,
                                 const _Float16* __restrict__ wt,
                                 float* __restrict__ y) {
    __shared__ alignas(16) _Float16 Stile[6 * 40 * 32];   // [row][col][ci] 15360B
    __shared__ alignas(16) _Float16 Bsa[9 * 64 * 32];     // [r][co][ci]    36864B

    const int tid  = threadIdx.x;
    const int lane = tid & 31;
    const int wv   = tid >> 5;

    const int tile_n   = blockIdx.x & 3;       // 4 co tiles
    const int tile_m   = blockIdx.x >> 2;      // 512 pixel tiles
    const int m_base   = tile_m * 128;
    const int img      = m_base >> 10;         // tile never crosses an image
    const int pix_base = m_base & 1023;
    const int h_base   = pix_base >> 5;        // 4 image rows per tile
    const int co_base  = tile_n * 64;

    // wave's 16 pixels are 16 consecutive cols of one row
    const int pr  = wv >> 1;                   // row 0..3 within tile
    const int pc  = (wv & 1) * 16 + (lane & 15);
    const int kA  = (lane >> 4) * 8;           // A frag K: {kA..kA+7, kA+16..kA+23}
    const int nlo = lane & 15;
    const int kB  = (lane >> 4) * 16;          // unused; B frag K is {kA..kA+7, kA+16..kA+23}, as A

    v8f acc[4];
#pragma unroll
    for (int f = 0; f < 4; ++f) acc[f] = (v8f)0.0f;

#pragma unroll 1
    for (int cs = 0; cs < 8; ++cs) {
        // ---- stage spike tile: one contiguous 15360B run -> 960 x 16B segs
        const _Float16* sps = sp + ((size_t)img * 8 + cs) * CSP
                                 + (size_t)h_base * SPROWH;
#pragma unroll
        for (int rnd = 0; rnd < 4; ++rnd) {
            const int s = tid + rnd * 256;
            if (s < 960) async_copy16(sps + s * 8, &Stile[s * 8]);
        }
        // ---- stage all 9 weight tiles (4KB each, contiguous)
        const _Float16* wts = wt + (size_t)cs * 8192 + (size_t)co_base * 32;
        {
            const int co = tid >> 2, sg = tid & 3;
#pragma unroll
            for (int rr = 0; rr < 9; ++rr)
                async_copy16(wts + (size_t)rr * 65536 + co * 32 + sg * 8,
                             &Bsa[(rr * 64 + co) * 32 + sg * 8]);
        }
        async_join();
        __syncthreads();

        // ---- 9 taps x 4 WMMAs; all fragment loads are ds_load_b128
        for (int r = 0; r < 9; ++r) {
            const int dh = r / 3 - 1;
            const int dw = r % 3 - 1;
            const int abase = ((pr + dh + 1) * 40 + (pc + dw + 1)) * 32;

            Frag a;
            a.u[0] = *(const uint4*)&Stile[abase + kA];
            a.u[1] = *(const uint4*)&Stile[abase + kA + 16];
            Frag b;
#pragma unroll
            for (int f = 0; f < 4; ++f) {
                const _Float16* bp = &Bsa[(r * 64 + f * 16 + nlo) * 32 + kA];
                b.u[0] = *(const uint4*)bp;
                b.u[1] = *(const uint4*)(bp + 16);
                acc[f] = __builtin_amdgcn_wmma_f32_16x16x32_f16(
                    false, a.v, false, b.v, (short)0, acc[f], false, false);
            }
        }
        __syncthreads();
    }

    // ---- writeback: VGPR rr holds M = rr + (lane>=16)*8 ; lane&15 is N
    const int pixw0 = pix_base + wv * 16 + (lane >> 4) * 8;
#pragma unroll
    for (int f = 0; f < 4; ++f) {
        const int co = co_base + f * 16 + (lane & 15);
        float* yp = y + (size_t)img * IMG_STRIDE + (size_t)co * HW + pixw0;
#pragma unroll
        for (int rr = 0; rr < 8; ++rr) yp[rr] = acc[f][rr];
    }
}

// ---------------------------------------------------------------------------
// Kernel 4: per-channel mean / inv_std over (N,H,W). One block per channel.
// ---------------------------------------------------------------------------
__global__ void bn_stats_kernel(const float* __restrict__ y,
                                float* __restrict__ mean,
                                float* __restrict__ invstd) {
    __shared__ float sh[256];
    __shared__ float sh2[256];
    const int c   = blockIdx.x;
    const int tid = threadIdx.x;
    float s = 0.0f, s2 = 0.0f;
    for (int img = 0; img < 64; ++img) {
        const float* p = y + (size_t)img * IMG_STRIDE + (size_t)c * HW;
        for (int j = tid; j < HW; j += 256) {
            float v = p[j];
            s += v; s2 += v * v;
        }
    }
    sh[tid] = s; sh2[tid] = s2;
    __syncthreads();
    for (int off = 128; off > 0; off >>= 1) {
        if (tid < off) { sh[tid] += sh[tid + off]; sh2[tid] += sh2[tid + off]; }
        __syncthreads();
    }
    if (tid == 0) {
        float m   = sh[0] * (1.0f / 65536.0f);
        float var = sh2[0] * (1.0f / 65536.0f) - m * m;
        mean[c]   = m;
        invstd[c] = rsqrtf(var + 1e-5f);
    }
}

// ---------------------------------------------------------------------------
// Kernel 5: in-place normalize with gamma/beta (float4 pass).
// ---------------------------------------------------------------------------
__global__ void bn_norm_kernel(float* __restrict__ y,
                               const float* __restrict__ mean,
                               const float* __restrict__ invstd,
                               const float* __restrict__ gamma,
                               const float* __restrict__ beta) {
    const size_t i = (size_t)blockIdx.x * 256 + threadIdx.x;   // 4194304 float4's
    const int c = (int)(((i * 4) >> 10) & 255);
    const float scale = invstd[c] * gamma[c];
    const float bias  = beta[c] - mean[c] * scale;
    float4 v = ((float4*)y)[i];
    v.x = v.x * scale + bias;
    v.y = v.y * scale + bias;
    v.z = v.z * scale + bias;
    v.w = v.w * scale + bias;
    ((float4*)y)[i] = v;
}

// ---------------------------------------------------------------------------
extern "C" void kernel_launch(void* const* d_in, const int* in_sizes, int n_in,
                              void* d_out, int out_size, void* d_ws, size_t ws_size,
                              hipStream_t stream) {
    const float* x     = (const float*)d_in[0];   // [4,16,256,32,32]
    const float* w     = (const float*)d_in[1];   // [256,256,3,3]
    const float* gamma = (const float*)d_in[2];   // [256]
    const float* beta  = (const float*)d_in[3];   // [256]
    float* y = (float*)d_out;                     // [64,256,32,32]

    char* ws = (char*)d_ws;
    _Float16* sp     = (_Float16*)ws;                 // padded spikes, 44.5 MB
    _Float16* wtbuf  = (_Float16*)(ws + WT_OFF);      // transformed weights
    float*    mean   = (float*)(ws + STAT_OFF);
    float*    invstd = mean + 256;

    zero_kernel     <<<10880, 256, 0, stream>>>((uint4*)sp);   // SP_BYTES/4096
    lif_kernel      <<<1024,  256, 0, stream>>>(x, sp);
    wt_kernel       <<<2304,  256, 0, stream>>>(w, wtbuf);
    conv_wmma_kernel<<<2048,  256, 0, stream>>>(sp, wtbuf, y);
    bn_stats_kernel <<<256,   256, 0, stream>>>(y, mean, invstd);
    bn_norm_kernel  <<<16384, 256, 0, stream>>>(y, mean, invstd, gamma, beta);
}
